// SpatioTemporalEncoder_17746804867601
// MI455X (gfx1250) — matrix operand order repaired and checked by reading, not compiled, not run on hardware
//
#include <hip/hip_runtime.h>
#include <hip/hip_bf16.h>

// ---------------------------------------------------------------------------
// Types for CDNA5 WMMA (gfx1250, wave32)
// ---------------------------------------------------------------------------
typedef __attribute__((ext_vector_type(16))) __bf16 v16bf;
typedef __attribute__((ext_vector_type(8)))  float  v8f;

struct Frag32B { uint4 lo, hi; };

__device__ __forceinline__ v16bf load_frag16(const unsigned short* p0,
                                             const unsigned short* p1) {
  Frag32B t;
  t.lo = *(const uint4*)p0;
  t.hi = *(const uint4*)p1;
  return __builtin_bit_cast(v16bf, t);
}

__device__ __forceinline__ unsigned short f2bf(float f) {
  unsigned u = __float_as_uint(f);
  u += 0x7FFFu + ((u >> 16) & 1u);          // round-to-nearest-even
  return (unsigned short)(u >> 16);
}

// ---------------------------------------------------------------------------
// bf16 WMMA GEMM: C[M,N] = act(A[M,K] @ B[K,N] + bias) + residual
//
// Block (256 thr = 8 waves) computes a 128x64 macro-tile:
//   - the 64-wide B panel (all K chunks) is staged once into LDS and shared
//     by all 8 waves (ds_load_b128 reads, ~8x less global B traffic)
//   - each wave owns 16 rows; one A fragment feeds 4 WMMAs (4 N-tiles);
//     all 4 B fragments are made live before the WMMA sequence so the
//     ds_loads clause together and the WMMAs issue back-to-back
// A: row-major bf16, K multiple of 32. Bp: pre-packed fragments with layout
// ((kc*Ntiles + nt)*32 + lane)*16 matching the v_wmma B lane layout.
// Requires M % 128 == 0, N % 64 == 0 (true for every GEMM in this model).
// act: 0=none, 1=exact GELU.
// ---------------------------------------------------------------------------
__global__ __launch_bounds__(256)
void gemm_bf16_wmma(const unsigned short* __restrict__ A,
                    const unsigned short* __restrict__ Bp,
                    const float* __restrict__ bias,
                    const float* __restrict__ resid,
                    float* __restrict__ Cf,
                    unsigned short* __restrict__ Cb,
                    int M, int N, int K, int act) {
  // LDS B panel: up to 8 K-chunks (K<=256) * 4 N-tiles * 32 lanes * 16 bf16
  __shared__ unsigned short Bsh[8 * 4 * 512];

  int ntn  = N >> 4;                 // total 16-wide N tiles
  int ng   = N >> 6;                 // 64-wide column groups
  int brow = blockIdx.x / ng;        // 128-row macro block
  int bcol = blockIdx.x - brow * ng; // 64-col group
  int kchunks = K >> 5;
  int tid  = threadIdx.x;

  // --- cooperative stage of the B panel into LDS (uint4 granularity) ---
  // global: ((kc*ntn + bcol*4 + j)*32 + lane)*16 ; lds: ((kc*4 + j)*32 + lane)*16
  int totalv = kchunks * 256;        // uint4 count = kchunks*4*32*2
  for (int i = tid; i < totalv; i += 256) {
    int half  = i & 1;
    int lane2 = (i >> 1) & 31;
    int j     = (i >> 6) & 3;
    int kc    = i >> 8;
    const uint4* src = (const uint4*)(Bp +
        (((size_t)(kc * ntn + bcol * 4 + j) * 32) + lane2) * 16) + half;
    *((uint4*)(Bsh + (((kc * 4 + j) * 32) + lane2) * 16) + half) = *src;
  }
  __syncthreads();

  int wave = tid >> 5;
  int lane = tid & 31;
  // A fragment: lanes 0-15 rows M=0..15 (K 0..7 / 16..23), lanes 16-31 same
  // rows (K 8..15 / 24..31).
  const unsigned short* Arow =
      A + (size_t)(brow * 128 + wave * 16 + (lane & 15)) * K + ((lane >> 4) << 3);

  v8f acc[4] = {};
  for (int kc = 0; kc < kchunks; ++kc) {
    if (kc + 1 < kchunks)
      __builtin_prefetch(Arow + (kc + 1) * 32, 0, 1);   // global_prefetch path
    v16bf a = load_frag16(Arow + kc * 32, Arow + kc * 32 + 16);
    const unsigned short* bs = Bsh + ((size_t)(kc * 4) * 32 + lane) * 16;
    // Make all 4 B fragments live before the WMMA burst: lets the compiler
    // clause the 8 ds_load_b128s, take one s_wait_dscnt, then issue the 4
    // v_wmma back-to-back on the XDL pipe.
    v16bf b0 = load_frag16(bs +    0, bs +    8);
    v16bf b1 = load_frag16(bs +  512, bs +  520);
    v16bf b2 = load_frag16(bs + 1024, bs + 1032);
    v16bf b3 = load_frag16(bs + 1536, bs + 1544);
    acc[0] = __builtin_amdgcn_wmma_f32_16x16x32_bf16(
        false, a, false, b0, (short)0, acc[0], false, false);
    acc[1] = __builtin_amdgcn_wmma_f32_16x16x32_bf16(
        false, a, false, b1, (short)0, acc[1], false, false);
    acc[2] = __builtin_amdgcn_wmma_f32_16x16x32_bf16(
        false, a, false, b2, (short)0, acc[2], false, false);
    acc[3] = __builtin_amdgcn_wmma_f32_16x16x32_bf16(
        false, a, false, b3, (short)0, acc[3], false, false);
  }

  int row0 = brow * 128 + wave * 16 + ((lane >> 4) << 3);  // VGPR r -> row0 + r
  int colb = bcol * 64 + (lane & 15);
#pragma unroll
  for (int j = 0; j < 4; ++j) {
    int col = colb + j * 16;
    float bi = bias ? bias[col] : 0.0f;
#pragma unroll
    for (int r = 0; r < 8; ++r) {
      float v = acc[j][r] + bi;
      if (act == 1) v = 0.5f * v * (1.0f + erff(v * 0.70710678118654752f));
      size_t o = (size_t)(row0 + r) * N + col;
      if (resid) v += resid[o];
      if (Cf) Cf[o] = v;
      else    Cb[o] = f2bf(v);
    }
  }
}

// ---------------------------------------------------------------------------
// Pack weight matrix into WMMA B-fragment order (bf16), zero-padding K.
// trans=0: W[k*N+n]; trans=1: W stored [N,Ksrc] -> B[k][n] = W[n*Ksrc+k].
// ---------------------------------------------------------------------------
__global__ void pack_b_frag(const float* __restrict__ W,
                            unsigned short* __restrict__ Bp,
                            int Ksrc, int Kpad, int N, int trans) {
  int idx = blockIdx.x * blockDim.x + threadIdx.x;
  int total = (Kpad >> 5) * (N >> 4) * 512;
  if (idx >= total) return;
  int e    = idx & 15;
  int lane = (idx >> 4) & 31;
  int t    = idx >> 9;
  int ntn  = N >> 4;
  int nt   = t % ntn;
  int kc   = t / ntn;
  int k = (kc << 5) + ((e >> 3) << 4) + ((lane >> 4) << 3) + (e & 7);  // lanes 0-15: K 0..7,16..23; 16-31: 8..15,24..31
  int n = (nt << 4) + (lane & 15);
  float w = 0.0f;
  if (k < Ksrc) w = trans ? W[(size_t)n * Ksrc + k] : W[(size_t)k * N + n];
  Bp[idx] = f2bf(w);
}

// f32 row-major [M,Ksrc] -> bf16 row-major [M,Kpad] (zero pad tail of K)
__global__ void cvt_f32_bf16_pad(const float* __restrict__ src,
                                 unsigned short* __restrict__ dst,
                                 long total, int Ksrc, int Kpad) {
  long idx = blockIdx.x * (long)blockDim.x + threadIdx.x;
  if (idx >= total) return;
  long m = idx / Kpad;
  int  k = (int)(idx - m * Kpad);
  dst[idx] = (k < Ksrc) ? f2bf(src[m * Ksrc + k]) : (unsigned short)0;
}

// ---------------------------------------------------------------------------
// GAT attention + aggregation + bias + BatchNorm(eval) + ELU.
// Chain topology (J=32) + self loop: dst j receives from {j-1, j+1, j}.
// One thread per (node, head).
// ---------------------------------------------------------------------------
__device__ __forceinline__ float dotC(const float* __restrict__ p,
                                      const float* __restrict__ a, int C) {
  float s = 0.f;
  for (int c = 0; c < C; ++c) s += p[c] * a[c];
  return s;
}

__global__ void gat_agg_bn_elu(const float* __restrict__ hw,
                               const float* __restrict__ asrc,
                               const float* __restrict__ adst,
                               const float* __restrict__ gbias,
                               const float* __restrict__ bng,
                               const float* __restrict__ bnb,
                               const float* __restrict__ bnm,
                               const float* __restrict__ bnv,
                               float* __restrict__ out,
                               int Nn, int H, int C) {
  int idx = blockIdx.x * blockDim.x + threadIdx.x;
  if (idx >= Nn * H) return;
  int n = idx / H;
  int h = idx - n * H;
  int j = n & 31;
  const float* av = asrc + h * C;
  const float* dv = adst + h * C;

  float ad_self = dotC(hw + (size_t)n * 64 + h * C, dv, C);

  int   srcs[3];
  float e[3];
  int   cnt = 0;
  srcs[cnt] = n;
  e[cnt++]  = dotC(hw + (size_t)n * 64 + h * C, av, C);
  if (j > 0)  { srcs[cnt] = n - 1; e[cnt++] = dotC(hw + (size_t)(n - 1) * 64 + h * C, av, C); }
  if (j < 31) { srcs[cnt] = n + 1; e[cnt++] = dotC(hw + (size_t)(n + 1) * 64 + h * C, av, C); }

  float m = -1e30f;
  for (int s = 0; s < cnt; ++s) {
    float t = e[s] + ad_self;
    e[s] = t > 0.f ? t : 0.2f * t;           // leaky_relu(0.2)
    m = fmaxf(m, e[s]);
  }
  float w[3], den = 0.f;
  for (int s = 0; s < cnt; ++s) { w[s] = __expf(e[s] - m); den += w[s]; }
  float inv = 1.f / den;

  for (int c = 0; c < C; ++c) {
    float acc = 0.f;
    for (int s = 0; s < cnt; ++s)
      acc += w[s] * hw[(size_t)srcs[s] * 64 + h * C + c];
    int cc = h * C + c;
    float v = acc * inv + gbias[cc];
    v = (v - bnm[cc]) * rsqrtf(bnv[cc] + 1e-5f) * bng[cc] + bnb[cc];
    out[(size_t)n * 64 + cc] = v > 0.f ? v : (__expf(v) - 1.f);   // ELU
  }
}

// Mean over J=32 joints: [G,32,64] f32 -> [G,64] bf16
__global__ void pool_joints(const float* __restrict__ h,
                            unsigned short* __restrict__ out, int G) {
  int idx = blockIdx.x * blockDim.x + threadIdx.x;
  if (idx >= G * 64) return;
  int g = idx >> 6, c = idx & 63;
  const float* p = h + ((size_t)g << 5) * 64 + c;
  float s = 0.f;
#pragma unroll
  for (int jj = 0; jj < 32; ++jj) s += p[jj * 64];
  out[idx] = f2bf(s * 0.03125f);
}

// seq[b,s,c] = (s==0 ? cls[c] : proj[b*60+s-1,c]) + posenc(s,c)
__global__ void assemble_seq(const float* __restrict__ proj,
                             const float* __restrict__ cls,
                             float* __restrict__ seq, int B) {
  int idx = blockIdx.x * blockDim.x + threadIdx.x;
  if (idx >= B * 61 * 128) return;
  int c  = idx & 127;
  int bs = idx >> 7;
  int s  = bs % 61;
  int b  = bs / 61;
  float base = (s == 0) ? cls[c] : proj[((size_t)b * 60 + (s - 1)) * 128 + c];
  float div  = __expf((float)(c & ~1) * -0.0719557842f);   // -ln(10000)/128
  float ang  = (float)s * div;
  seq[idx] = base + ((c & 1) ? __cosf(ang) : __sinf(ang));
}

// LayerNorm(D=128), one wave per row, bf16 output
__global__ void ln_rows_bf16(const float* __restrict__ X,
                             const float* __restrict__ g,
                             const float* __restrict__ b,
                             unsigned short* __restrict__ out, int rows) {
  int wave = (blockIdx.x * blockDim.x + threadIdx.x) >> 5;
  int lane = threadIdx.x & 31;
  if (wave >= rows) return;
  const float* x = X + (size_t)wave * 128 + lane * 4;
  float4 v = *(const float4*)x;
  float s = v.x + v.y + v.z + v.w;
  float q = v.x * v.x + v.y * v.y + v.z * v.z + v.w * v.w;
#pragma unroll
  for (int m = 16; m >= 1; m >>= 1) {
    s += __shfl_xor(s, m, 32);
    q += __shfl_xor(q, m, 32);
  }
  float mean = s * (1.0f / 128.0f);
  float rstd = rsqrtf(q * (1.0f / 128.0f) - mean * mean + 1e-5f);
  int c = lane * 4;
  unsigned short* o = out + (size_t)wave * 128 + c;
  o[0] = f2bf((v.x - mean) * rstd * g[c + 0] + b[c + 0]);
  o[1] = f2bf((v.y - mean) * rstd * g[c + 1] + b[c + 1]);
  o[2] = f2bf((v.z - mean) * rstd * g[c + 2] + b[c + 2]);
  o[3] = f2bf((v.w - mean) * rstd * g[c + 3] + b[c + 3]);
}

// Final LayerNorm of CLS row (s=0) -> f32 output [B,128]
__global__ void ln_cls_out(const float* __restrict__ seq,
                           const float* __restrict__ g,
                           const float* __restrict__ b,
                           float* __restrict__ out, int B) {
  int wave = (blockIdx.x * blockDim.x + threadIdx.x) >> 5;
  int lane = threadIdx.x & 31;
  if (wave >= B) return;
  const float* x = seq + (size_t)wave * 61 * 128 + lane * 4;
  float4 v = *(const float4*)x;
  float s = v.x + v.y + v.z + v.w;
  float q = v.x * v.x + v.y * v.y + v.z * v.z + v.w * v.w;
#pragma unroll
  for (int m = 16; m >= 1; m >>= 1) {
    s += __shfl_xor(s, m, 32);
    q += __shfl_xor(q, m, 32);
  }
  float mean = s * (1.0f / 128.0f);
  float rstd = rsqrtf(q * (1.0f / 128.0f) - mean * mean + 1e-5f);
  int c = lane * 4;
  float* o = out + (size_t)wave * 128 + c;
  o[0] = (v.x - mean) * rstd * g[c + 0] + b[c + 0];
  o[1] = (v.y - mean) * rstd * g[c + 1] + b[c + 1];
  o[2] = (v.z - mean) * rstd * g[c + 2] + b[c + 2];
  o[3] = (v.w - mean) * rstd * g[c + 3] + b[c + 3];
}

// MHA core: one block per (batch, head); S=61, dh=32; Q/K/V staged in LDS.
__global__ __launch_bounds__(64)
void mha_attn(const float* __restrict__ qkv,
              unsigned short* __restrict__ aout, int B) {
  __shared__ float Q[61][32];
  __shared__ float Ks[61][32];
  __shared__ float Vs[61][32];
  int b   = blockIdx.x >> 2;
  int h   = blockIdx.x & 3;
  int tid = threadIdx.x;
  for (int i = tid; i < 61 * 32; i += 64) {
    int s = i >> 5, c = i & 31;
    const float* row = qkv + (size_t)(b * 61 + s) * 384 + h * 32 + c;
    Q[s][c]  = row[0];
    Ks[s][c] = row[128];
    Vs[s][c] = row[256];
  }
  __syncthreads();
  int q = tid;
  if (q < 61) {
    const float scale = 0.1767766952966369f;   // 1/sqrt(32)
    float m = -1e30f;
    for (int k = 0; k < 61; ++k) {
      float d = 0.f;
#pragma unroll
      for (int c = 0; c < 32; ++c) d += Q[q][c] * Ks[k][c];
      m = fmaxf(m, d * scale);
    }
    float acc[32];
#pragma unroll
    for (int c = 0; c < 32; ++c) acc[c] = 0.f;
    float den = 0.f;
    for (int k = 0; k < 61; ++k) {
      float d = 0.f;
#pragma unroll
      for (int c = 0; c < 32; ++c) d += Q[q][c] * Ks[k][c];
      float w = __expf(d * scale - m);
      den += w;
#pragma unroll
      for (int c = 0; c < 32; ++c) acc[c] += w * Vs[k][c];
    }
    float inv = 1.f / den;
    unsigned short* o = aout + (size_t)(b * 61 + q) * 128 + h * 32;
#pragma unroll
    for (int c = 0; c < 32; ++c) o[c] = f2bf(acc[c] * inv);
  }
}

// ---------------------------------------------------------------------------
// Host: orchestration.  Input order assumes jax tree flattening (sorted keys).
// ---------------------------------------------------------------------------
extern "C" void kernel_launch(void* const* d_in, const int* in_sizes, int n_in,
                              void* d_out, int out_size, void* d_ws, size_t ws_size,
                              hipStream_t stream) {
  (void)in_sizes; (void)n_in; (void)out_size; (void)ws_size;

  const int BATCH = 256, T = 60, FIN = 9;
  const int NN = BATCH * T * 32;      // 491520 nodes (NN/128 = 3840)
  const int G  = BATCH * T;           // 15360 graphs (G/128 = 120)
  const int ROWS = BATCH * (T + 1);   // 15616 sequence rows (ROWS/128 = 122)
  const int D = 128;

  // --- inputs (sorted-key flatten order) ---
  const float* bnp[3][4];   // [layer][b,g,mean,var]
  for (int i = 0; i < 3; ++i)
    for (int k = 0; k < 4; ++k) bnp[i][k] = (const float*)d_in[1 + i * 4 + k];
  const float* cls = (const float*)d_in[13];
  const float *gatW[3], *gatAd[3], *gatAs[3], *gatB[3];
  for (int i = 0; i < 3; ++i) {
    int b0 = 14 + i * 4;
    gatW[i]  = (const float*)d_in[b0 + 0];
    gatAd[i] = (const float*)d_in[b0 + 1];
    gatAs[i] = (const float*)d_in[b0 + 2];
    gatB[i]  = (const float*)d_in[b0 + 3];
  }
  // layers: W1,W2,Win,Wout,b1,b2,bin,bout,ln1b,ln1g,ln2b,ln2g
  const float* L[2][12];
  for (int l = 0; l < 2; ++l)
    for (int k = 0; k < 12; ++k) L[l][k] = (const float*)d_in[26 + l * 12 + k];
  const float* normb = (const float*)d_in[50];
  const float* normg = (const float*)d_in[51];
  const float* projW = (const float*)d_in[52];
  const float* projb = (const float*)d_in[53];
  const float* x     = (const float*)d_in[54];

  // --- workspace carve ---
  char* wp = (char*)d_ws;
  auto carve = [&](size_t bytes) -> void* {
    void* r = (void*)wp;
    wp += (bytes + 255) & ~(size_t)255;
    return r;
  };
  unsigned short* bfA    = (unsigned short*)carve((size_t)NN * 64 * 2);
  float*          hw     = (float*)carve((size_t)NN * 64 * 4);
  float*          hA     = (float*)carve((size_t)NN * 64 * 4);
  unsigned short* pooled = (unsigned short*)carve((size_t)G * 64 * 2);
  float*          projO  = (float*)carve((size_t)G * D * 4);
  float*          seq    = (float*)carve((size_t)ROWS * D * 4);
  unsigned short* lnbuf  = (unsigned short*)carve((size_t)ROWS * D * 2);
  float*          qkvb   = (float*)carve((size_t)ROWS * 384 * 4);
  unsigned short* attnb  = (unsigned short*)carve((size_t)ROWS * D * 2);
  unsigned short* ffnb   = (unsigned short*)carve((size_t)ROWS * 256 * 2);
  unsigned short* pGat[3];
  pGat[0] = (unsigned short*)carve(32 * 64 * 2);
  pGat[1] = (unsigned short*)carve(64 * 64 * 2);
  pGat[2] = (unsigned short*)carve(64 * 64 * 2);
  unsigned short* pProj = (unsigned short*)carve(64 * 128 * 2);
  unsigned short *pQKV[2], *pOut[2], *pF1[2], *pF2[2];
  for (int l = 0; l < 2; ++l) {
    pQKV[l] = (unsigned short*)carve(128 * 384 * 2);
    pOut[l] = (unsigned short*)carve(128 * 128 * 2);
    pF1[l]  = (unsigned short*)carve(128 * 256 * 2);
    pF2[l]  = (unsigned short*)carve(256 * 128 * 2);
  }

  auto nb = [](long total, int bs) { return (unsigned)((total + bs - 1) / bs); };
  auto gemm = [&](const unsigned short* A, const unsigned short* Bp,
                  const float* bias, const float* resid,
                  float* Cf, unsigned short* Cb, int M, int N, int K, int act) {
    unsigned blocks = (unsigned)((M >> 7) * (N >> 6));   // 128x64 macro-tiles
    gemm_bf16_wmma<<<blocks, 256, 0, stream>>>(A, Bp, bias, resid, Cf, Cb,
                                               M, N, K, act);
  };
  auto packB = [&](const float* W, unsigned short* Bp, int Ksrc, int Kpad,
                   int N, int trans) {
    long tot = (long)(Kpad >> 5) * (N >> 4) * 512;
    pack_b_frag<<<nb(tot, 256), 256, 0, stream>>>(W, Bp, Ksrc, Kpad, N, trans);
  };

  // --- pack all weights (deterministic, every call) ---
  packB(gatW[0], pGat[0], FIN, 32, 64, 0);
  packB(gatW[1], pGat[1], 64, 64, 64, 0);
  packB(gatW[2], pGat[2], 64, 64, 64, 0);
  packB(projW, pProj, 64, 64, 128, 0);
  for (int l = 0; l < 2; ++l) {
    packB(L[l][2], pQKV[l], 128, 128, 384, 1);   // Win  [384,128] -> B=Win^T
    packB(L[l][3], pOut[l], 128, 128, 128, 1);   // Wout [128,128]
    packB(L[l][0], pF1[l],  128, 128, 256, 1);   // W1   [256,128]
    packB(L[l][1], pF2[l],  256, 256, 128, 1);   // W2   [128,256]
  }

  // --- GAT stack ---
  cvt_f32_bf16_pad<<<nb((long)NN * 32, 256), 256, 0, stream>>>(x, bfA,
                                                               (long)NN * 32, FIN, 32);
  gemm(bfA, pGat[0], nullptr, nullptr, hw, nullptr, NN, 64, 32, 0);
  gat_agg_bn_elu<<<nb((long)NN * 4, 256), 256, 0, stream>>>(
      hw, gatAs[0], gatAd[0], gatB[0], bnp[0][1], bnp[0][0], bnp[0][2], bnp[0][3],
      hA, NN, 4, 16);

  cvt_f32_bf16_pad<<<nb((long)NN * 64, 256), 256, 0, stream>>>(hA, bfA,
                                                               (long)NN * 64, 64, 64);
  gemm(bfA, pGat[1], nullptr, nullptr, hw, nullptr, NN, 64, 64, 0);
  gat_agg_bn_elu<<<nb((long)NN * 4, 256), 256, 0, stream>>>(
      hw, gatAs[1], gatAd[1], gatB[1], bnp[1][1], bnp[1][0], bnp[1][2], bnp[1][3],
      hA, NN, 4, 16);

  cvt_f32_bf16_pad<<<nb((long)NN * 64, 256), 256, 0, stream>>>(hA, bfA,
                                                               (long)NN * 64, 64, 64);
  gemm(bfA, pGat[2], nullptr, nullptr, hw, nullptr, NN, 64, 64, 0);
  gat_agg_bn_elu<<<nb((long)NN, 256), 256, 0, stream>>>(
      hw, gatAs[2], gatAd[2], gatB[2], bnp[2][1], bnp[2][0], bnp[2][2], bnp[2][3],
      hA, NN, 1, 64);

  // --- pool + projection + sequence assembly ---
  pool_joints<<<nb((long)G * 64, 256), 256, 0, stream>>>(hA, pooled, G);
  gemm(pooled, pProj, projb, nullptr, projO, nullptr, G, 128, 64, 0);
  assemble_seq<<<nb((long)BATCH * 61 * 128, 256), 256, 0, stream>>>(projO, cls,
                                                                    seq, BATCH);

  // --- transformer encoder layers (norm_first) ---
  for (int l = 0; l < 2; ++l) {
    ln_rows_bf16<<<nb((long)ROWS * 32, 256), 256, 0, stream>>>(
        seq, L[l][9], L[l][8], lnbuf, ROWS);                        // ln1g, ln1b
    gemm(lnbuf, pQKV[l], L[l][6], nullptr, qkvb, nullptr, ROWS, 384, 128, 0);
    mha_attn<<<BATCH * 4, 64, 0, stream>>>(qkvb, attnb, BATCH);
    gemm(attnb, pOut[l], L[l][7], seq, seq, nullptr, ROWS, 128, 128, 0);
    ln_rows_bf16<<<nb((long)ROWS * 32, 256), 256, 0, stream>>>(
        seq, L[l][11], L[l][10], lnbuf, ROWS);                      // ln2g, ln2b
    gemm(lnbuf, pF1[l], L[l][4], nullptr, nullptr, ffnb, ROWS, 256, 128, 1);
    gemm(ffnb, pF2[l], L[l][5], seq, seq, nullptr, ROWS, 128, 256, 0);
  }

  // --- final LN of CLS row ---
  ln_cls_out<<<nb((long)BATCH * 32, 256), 256, 0, stream>>>(seq, normg, normb,
                                                            (float*)d_out, BATCH);
}
